// OuterProductMean_66425964200174
// MI455X (gfx1250) — matrix operand order repaired and checked by reading, not compiled, not run on hardware
//
#include <hip/hip_runtime.h>

#define C_M   256
#define C_H   32
#define C_Z   128
#define N_SEQ 128
#define N_RES 256

typedef __attribute__((ext_vector_type(16))) __bf16 v16bf;
typedef __attribute__((ext_vector_type(8)))  float  v8f;

union BF16x16 { v16bf v; uint4 q[2]; };

// Load one lane's 32-byte slice of a pre-swizzled WMMA operand tile
// (tile = 1024 contiguous bytes per wave, 32B per lane -> 2x global_load_b128).
__device__ inline v16bf load_tile_g(const __bf16* __restrict__ tile, int lane) {
    BF16x16 t;
    const uint4* q = (const uint4*)(tile + lane * 16);
    t.q[0] = q[0];
    t.q[1] = q[1];
    return t.v;
}

// ---------------------------------------------------------------------------
// Kernel 1: LayerNorm + a/b projections, packed as bf16 into WMMA operand
// layouts.  aW: A-operand layout (M = a-channel c, K = s).
//           bW: B-operand layout (K = s, N = b-channel e).
// aW/bW shape: [r][tile2][kt(4)][lane(32)][elem(16)]  (512 bf16 per tile)
// ---------------------------------------------------------------------------
__global__ __launch_bounds__(256) void opm_prep(
    const float* __restrict__ m, const float* __restrict__ mask,
    const float* __restrict__ gamma, const float* __restrict__ beta,
    const float* __restrict__ W1, const float* __restrict__ b1,
    const float* __restrict__ W2, const float* __restrict__ b2,
    __bf16* __restrict__ aW, __bf16* __restrict__ bW)
{
    __shared__ float lnbuf[8][C_M];
    const int wave = threadIdx.x >> 5;
    const int lane = threadIdx.x & 31;
    const int row  = blockIdx.x * 8 + wave;   // row = s * N_RES + r
    const int s = row >> 8;
    const int r = row & (N_RES - 1);

    const float* mrow = m + (size_t)row * C_M;
    float x[8], sum = 0.f, sumsq = 0.f;
#pragma unroll
    for (int i = 0; i < 8; ++i) {
        x[i] = mrow[lane + 32 * i];
        sum += x[i];
        sumsq += x[i] * x[i];
    }
#pragma unroll
    for (int off = 16; off > 0; off >>= 1) {   // wave32 reduction
        sum   += __shfl_xor(sum, off, 32);
        sumsq += __shfl_xor(sumsq, off, 32);
    }
    const float mu  = sum * (1.f / C_M);
    const float var = sumsq * (1.f / C_M) - mu * mu;
    const float rstd = rsqrtf(var + 1e-5f);
#pragma unroll
    for (int i = 0; i < 8; ++i) {
        const int c = lane + 32 * i;
        lnbuf[wave][c] = (x[i] - mu) * rstd * gamma[c] + beta[c];
    }
    __syncthreads();

    const int h = lane;                        // one hidden channel per lane
    float a1 = b1[h], a2 = b2[h];
#pragma unroll 4
    for (int c = 0; c < C_M; ++c) {
        const float l = lnbuf[wave][c];
        a1 += l * W1[c * C_H + h];
        a2 += l * W2[c * C_H + h];
    }
    const float mk = mask[row];
    a1 *= mk;
    a2 *= mk;

    const int k  = s & 31;
    const int kt = s >> 5;
    // --- pack a into A-operand layout (M = h, K = s) ---
    {
        const int mt = h >> 4, mm = h & 15;
        int khalf, e;
        if      (k < 8)  { khalf = 0; e = k;      }
        else if (k < 16) { khalf = 1; e = k - 8;  }
        else if (k < 24) { khalf = 0; e = k - 8;  }
        else             { khalf = 1; e = k - 16; }
        const size_t idx =
            ((((size_t)r * 2 + mt) * 4 + kt) * 32 + (khalf * 16 + mm)) * 16 + e;
        aW[idx] = (__bf16)a1;
    }
    // --- pack b into B-operand layout (K = s, N = h) ---
    {
        const int nt = h >> 4, nn = h & 15;
        const int khalf = (k >> 3) & 1, ee = (k & 7) | ((k >> 4) << 3);
        const size_t idx =
            ((((size_t)r * 2 + nt) * 4 + kt) * 32 + (khalf * 16 + nn)) * 16 + ee;
        bW[idx] = (__bf16)a2;
    }
}

// ---------------------------------------------------------------------------
// Kernel 2: Wout [1024][128] f32 -> bf16 in B-operand layout
// WoutW shape: [kt(32)][ntile(8)][lane(32)][elem(16)]
// ---------------------------------------------------------------------------
__global__ __launch_bounds__(256) void opm_pack_wout(
    const float* __restrict__ Wout, __bf16* __restrict__ WoutW)
{
    const int t  = blockIdx.x * 256 + threadIdx.x;   // t = ce*128 + z
    const int ce = t >> 7, z = t & 127;
    const int kt = ce >> 5, k = ce & 31, khalf = (k >> 3) & 1, ee = (k & 7) | ((k >> 4) << 3);
    const int nt = z >> 4,  nn = z & 15;
    const size_t idx = (((size_t)kt * 8 + nt) * 32 + (khalf * 16 + nn)) * 16 + ee;
    WoutW[idx] = (__bf16)Wout[t];
}

// ---------------------------------------------------------------------------
// Kernel 3: invn[r1][r2] = 1 / (sum_s mask[s,r1]*mask[s,r2] + eps)
// ---------------------------------------------------------------------------
__global__ __launch_bounds__(256) void opm_masknorm(
    const float* __restrict__ mask, float* __restrict__ invn)
{
    const int r1 = blockIdx.x, r2 = threadIdx.x;
    float d = 0.f;
#pragma unroll 4
    for (int s = 0; s < N_SEQ; ++s)
        d += mask[s * N_RES + r1] * mask[s * N_RES + r2];
    invn[r1 * N_RES + r2] = 1.f / (d + 1e-3f);
}

// ---------------------------------------------------------------------------
// Kernel 4 (hot): fused outer-product GEMM + Wout GEMM.
// Workgroup = 8 waves = 16 (r1,r2) pairs (4x4 tile).
// Stage 1: wave w -> r1 = R1b + (w>>1), r2 = R2b + (w&1)*2 + {0,1};
//          two 32x32xK=128 bf16 WMMA GEMMs, result -> LDS V[16][1024] bf16.
// Stage 2: [16 pairs x 1024] x WoutW -> wave w owns z-tile w (16 columns).
// ---------------------------------------------------------------------------
#define VSTRIDE 1032   // 1024 + 8 bf16 pad -> 516 dwords, conflict-spreading

__global__ __launch_bounds__(256) void opm_main(
    const __bf16* __restrict__ aW, const __bf16* __restrict__ bW,
    const __bf16* __restrict__ WoutW, const float* __restrict__ bout,
    const float* __restrict__ invn, float* __restrict__ out)
{
    __shared__ __bf16 V[16 * VSTRIDE];

    const int wave = threadIdx.x >> 5;
    const int lane = threadIdx.x & 31;
    const int R1b  = blockIdx.y * 4;
    const int R2b  = blockIdx.x * 4;

    // ---- Stage 1: pair GEMMs ----
    const int i1  = wave >> 1;
    const int r1  = R1b + i1;
    const int i2a = (wave & 1) * 2;
    const int r2a = R2b + i2a;

    const __bf16* aT  = aW + (size_t)r1 * 4096;          // [mt(2)][kt(4)][512]
    const __bf16* bTa = bW + (size_t)r2a * 4096;
    const __bf16* bTb = bTa + 4096;                      // r2a + 1

    v8f accA[2][2] = {};  // pair (r1, r2a):   [mt][nt]
    v8f accB[2][2] = {};  // pair (r1, r2a+1)
#pragma unroll
    for (int kt = 0; kt < 4; ++kt) {
        const v16bf A0  = load_tile_g(aT  + (0 * 4 + kt) * 512, lane);
        const v16bf A1  = load_tile_g(aT  + (1 * 4 + kt) * 512, lane);
        const v16bf B0a = load_tile_g(bTa + (0 * 4 + kt) * 512, lane);
        const v16bf B1a = load_tile_g(bTa + (1 * 4 + kt) * 512, lane);
        const v16bf B0b = load_tile_g(bTb + (0 * 4 + kt) * 512, lane);
        const v16bf B1b = load_tile_g(bTb + (1 * 4 + kt) * 512, lane);
        accA[0][0] = __builtin_amdgcn_wmma_f32_16x16x32_bf16(false, A0, false, B0a, (short)0, accA[0][0], false, false);
        accA[0][1] = __builtin_amdgcn_wmma_f32_16x16x32_bf16(false, A0, false, B1a, (short)0, accA[0][1], false, false);
        accA[1][0] = __builtin_amdgcn_wmma_f32_16x16x32_bf16(false, A1, false, B0a, (short)0, accA[1][0], false, false);
        accA[1][1] = __builtin_amdgcn_wmma_f32_16x16x32_bf16(false, A1, false, B1a, (short)0, accA[1][1], false, false);
        accB[0][0] = __builtin_amdgcn_wmma_f32_16x16x32_bf16(false, A0, false, B0b, (short)0, accB[0][0], false, false);
        accB[0][1] = __builtin_amdgcn_wmma_f32_16x16x32_bf16(false, A0, false, B1b, (short)0, accB[0][1], false, false);
        accB[1][0] = __builtin_amdgcn_wmma_f32_16x16x32_bf16(false, A1, false, B0b, (short)0, accB[1][0], false, false);
        accB[1][1] = __builtin_amdgcn_wmma_f32_16x16x32_bf16(false, A1, false, B1b, (short)0, accB[1][1], false, false);
    }

    // spill pair results to LDS as bf16, V[p][c*32 + e]
    {
        const int pA  = i1 * 4 + i2a;
        const int pB  = pA + 1;
        const int nlo = lane & 15;
        const int mhi = (lane >> 4) * 8;   // C/D layout: M = (lane>=16)*8 + j
#pragma unroll
        for (int mt = 0; mt < 2; ++mt)
#pragma unroll
            for (int nt = 0; nt < 2; ++nt)
#pragma unroll
                for (int j = 0; j < 8; ++j) {
                    const int c = mt * 16 + mhi + j;
                    const int e = nt * 16 + nlo;
                    V[pA * VSTRIDE + c * 32 + e] = (__bf16)accA[mt][nt][j];
                    V[pB * VSTRIDE + c * 32 + e] = (__bf16)accB[mt][nt][j];
                }
    }
    __syncthreads();

    // ---- Stage 2: [16 x 1024] @ WoutW -> 16 z-columns per wave ----
    v8f acc = {};
    const __bf16* wB = WoutW + (size_t)wave * 512;   // ntile = wave
    const int prow = lane & 15;                      // A-operand M = pair id
    const int kh   = lane >> 4;
    const __bf16* vbase = &V[prow * VSTRIDE + kh * 8];
#pragma unroll 4
    for (int kt = 0; kt < 32; ++kt) {
        BF16x16 A;
        A.q[0] = *(const uint4*)(vbase + kt * 32);        // K-chunk 0..7
        A.q[1] = *(const uint4*)(vbase + kt * 32 + 16);   // K-chunk 16..23
        const v16bf B = load_tile_g(wB + (size_t)kt * 4096, lane);
        acc = __builtin_amdgcn_wmma_f32_16x16x32_bf16(false, A.v, false, B, (short)0, acc, false, false);
    }

    // ---- Epilogue: + bout, * invn, store f32 ----
    const int z  = wave * 16 + (lane & 15);
    const float bz = bout[z];
#pragma unroll
    for (int j = 0; j < 8; ++j) {
        const int p   = (lane >> 4) * 8 + j;
        const int rr1 = R1b + (p >> 2);
        const int rr2 = R2b + (p & 3);
        const float v = (acc[j] + bz) * invn[rr1 * N_RES + rr2];
        out[((size_t)rr1 * N_RES + rr2) * C_Z + z] = v;
    }
}

// ---------------------------------------------------------------------------
extern "C" void kernel_launch(void* const* d_in, const int* in_sizes, int n_in,
                              void* d_out, int out_size, void* d_ws, size_t ws_size,
                              hipStream_t stream) {
    const float* m     = (const float*)d_in[0];
    const float* mask  = (const float*)d_in[1];
    const float* gamma = (const float*)d_in[2];
    const float* beta  = (const float*)d_in[3];
    const float* W1    = (const float*)d_in[4];
    const float* b1    = (const float*)d_in[5];
    const float* W2    = (const float*)d_in[6];
    const float* b2    = (const float*)d_in[7];
    const float* Wout  = (const float*)d_in[8];
    const float* bout  = (const float*)d_in[9];
    float* out = (float*)d_out;

    // workspace layout (~4.5 MB, fully rewritten every call)
    char* ws = (char*)d_ws;
    __bf16* aW    = (__bf16*)(ws);                              // 2 MB
    __bf16* bW    = (__bf16*)(ws + (size_t)2 * 1024 * 1024);    // 2 MB
    __bf16* WoutW = (__bf16*)(ws + (size_t)4 * 1024 * 1024);    // 256 KB
    float*  invn  = (float*) (ws + (size_t)4 * 1024 * 1024 + 256 * 1024); // 256 KB

    opm_prep<<<(N_SEQ * N_RES) / 8, 256, 0, stream>>>(
        m, mask, gamma, beta, W1, b1, W2, b2, aW, bW);
    opm_pack_wout<<<(C_H * C_H * C_Z) / 256, 256, 0, stream>>>(Wout, WoutW);
    opm_masknorm<<<N_RES, N_RES, 0, stream>>>(mask, invn);
    opm_main<<<dim3(N_RES / 4, N_RES / 4), 256, 0, stream>>>(
        aW, bW, WoutW, bout, invn, out);
}
